// Decoder_45603962749635
// MI455X (gfx1250) — matrix operand order repaired and checked by reading, not compiled, not run on hardware
//
#include <hip/hip_runtime.h>

// Autoregressive tanh-RNN decode for MI455X (gfx1250, wave32).
// h = tanh(x*W_ih^T + b_ih + h*W_hh^T + b_hh); y = h*W_out^T + b_out (256 steps).
// Recurrence GEMM uses v_wmma_f32_16x16x32_bf16 with double-buffered
// GLOBAL_LOAD_ASYNC_TO_LDS_B128 staging (ASYNCcnt) of A(h) and B(W_hh) tiles.
// W_hh converted to bf16 once (8 MB, L2-resident @ 192 MB L2).

#define T_STEPS 256
#define B_SZ    512
#define H_SZ    2048

// GEMM tiling: WG = 256 thr = 8 waves (4 in M x 2 in N); wave tile 16x64.
// WG tile 64(M) x 128(N), K-step 64, 32 stages, double-buffered LDS.
#define LDS_ROW   144u                     // 64 bf16 = 128B, padded to 144B
#define A_ROWS    64u
#define B_ROWS    128u
#define A_BYTES   (A_ROWS * LDS_ROW)       // 9216
#define B_BYTES   (B_ROWS * LDS_ROW)       // 18432
#define BUF_BYTES (A_BYTES + B_BYTES)      // 27648
#define N_STAGES  (H_SZ / 64)              // 32

typedef __attribute__((ext_vector_type(16))) __bf16 bf16x16;
typedef __attribute__((ext_vector_type(8)))  __bf16 bf16x8;
typedef __attribute__((ext_vector_type(8)))  float  v8f;
typedef __attribute__((ext_vector_type(4)))  float  v4f;

// ---------------------------------------------------------------------------
// Prep: W_hh fp32 -> bf16, h0 fp32 -> bf16, bias_sum = b_ih + b_hh, x = x0.
// ---------------------------------------------------------------------------
__global__ void rnn_prep(const float* __restrict__ W_hh,
                         const float* __restrict__ h0,
                         const float* __restrict__ b_ih,
                         const float* __restrict__ b_hh,
                         const float* __restrict__ x0,
                         __bf16* __restrict__ Whh_bf,
                         __bf16* __restrict__ h_bf,
                         float* __restrict__ bias_sum,
                         float* __restrict__ x_cur) {
    const int i = blockIdx.x * blockDim.x + threadIdx.x;
    if (i < H_SZ * H_SZ) Whh_bf[i] = (__bf16)W_hh[i];
    if (i < B_SZ * H_SZ) h_bf[i]   = (__bf16)h0[i];
    if (i < H_SZ)        bias_sum[i] = b_ih[i] + b_hh[i];
    if (i < B_SZ)        x_cur[i]  = x0[i];
}

// ---------------------------------------------------------------------------
// One recurrence step: h_out = tanh(bias + x*W_ih^T + h_in @ W_hh^T).
// Grid (H/128, B/64) = (16, 8) -> 128 WGs. C[b,n] = sum_k h[b,k]*W_hh[n,k].
// ---------------------------------------------------------------------------
__global__ __launch_bounds__(256) void rnn_step_gemm(
    const __bf16* __restrict__ h_in,     // [B,H] bf16
    __bf16*       __restrict__ h_out,    // [B,H] bf16
    const float*  __restrict__ x_cur,    // [B]
    const __bf16* __restrict__ Whh,      // [H,H] bf16, row n = weights of unit n
    const float*  __restrict__ W_ih,     // [H]
    const float*  __restrict__ bias_sum) // [H]
{
    __shared__ __align__(16) char smem[2 * BUF_BYTES];   // 55296 B

    const int tid  = threadIdx.x;
    const int lane = tid & 31;
    const int wave = tid >> 5;
    const int wm   = wave >> 1;          // 0..3 : M quarter of WG tile
    const int wn   = wave & 1;           // 0..1 : N half of WG tile
    const int mBase = blockIdx.y * 64 + wm * 16;
    const int nBase = blockIdx.x * 128 + wn * 64;

    const int aKoff = (lane >> 4) << 3;  // 0 or 8  (A-frag K base, elems)
    const int bKoff = (lane >> 4) << 3;  // 0 or 8  (B-frag K base, elems)
    const int nCol  = lane & 15;

    // ---- async staging setup (6 b128 transfers / thread / stage) ----------
    const unsigned mWG = blockIdx.y * 64u;
    const unsigned nWG = blockIdx.x * 128u;
    const unsigned rowT   = (unsigned)(tid >> 3);        // 0..31
    const unsigned chunkB = (unsigned)(tid & 7) * 16u;   // 16B chunk in row
    // global byte offsets (row stride = H*2 = 4096B); + t*32 rows + kByte
    const unsigned aG0 = (mWG + rowT) * 4096u + chunkB;
    const unsigned bG0 = (nWG + rowT) * 4096u + chunkB;
    // LDS byte offsets within a buffer
    const unsigned aL0 = rowT * LDS_ROW + chunkB;
    const unsigned bL0 = A_BYTES + rowT * LDS_ROW + chunkB;
    const unsigned ldsBase0 = (unsigned)(size_t)(&smem[0]);

    auto issue_stage = [&](int ks, int buf) {
        const unsigned kByte = (unsigned)ks * 128u;      // 64 elems * 2B
        const unsigned lbase = ldsBase0 + (unsigned)buf * BUF_BYTES;
#pragma unroll
        for (unsigned t = 0; t < 2; ++t) {               // A tile: 64 rows
            asm volatile("global_load_async_to_lds_b128 %0, %1, %2"
                         :: "v"(lbase + aL0 + t * 32u * LDS_ROW),
                            "v"(aG0 + kByte + t * 32u * 4096u),
                            "s"(h_in)
                         : "memory");
        }
#pragma unroll
        for (unsigned t = 0; t < 4; ++t) {               // B tile: 128 rows
            asm volatile("global_load_async_to_lds_b128 %0, %1, %2"
                         :: "v"(lbase + bL0 + t * 32u * LDS_ROW),
                            "v"(bG0 + kByte + t * 32u * 4096u),
                            "s"(Whh)
                         : "memory");
        }
    };

    v8f acc[4] = {};

    issue_stage(0, 0);
    for (int ks = 0; ks < N_STAGES; ++ks) {
        const int cur = ks & 1;
        if (ks + 1 < N_STAGES) {
            issue_stage(ks + 1, cur ^ 1);
            // 6 outstanding = the prefetch; in-order completion => stage ks done
            asm volatile("s_wait_asynccnt 0x6" ::: "memory");
        } else {
            asm volatile("s_wait_asynccnt 0x0" ::: "memory");
        }
        __syncthreads();

        const char* bufc = smem + cur * BUF_BYTES;
        const char* aBase = bufc + (unsigned)(wm * 16 + (lane & 15)) * LDS_ROW;
        const char* bBase = bufc + A_BYTES + (unsigned)(wn * 64 + nCol) * LDS_ROW;

#pragma unroll
        for (int s2 = 0; s2 < 2; ++s2) {                 // two k=32 sub-steps
            // A-frag: elems 0..7 = K(aKoff..+7), elems 8..15 = K(aKoff+16..+23)
            bf16x16 af;
            ((bf16x8*)&af)[0] = *(const bf16x8*)(aBase + s2 * 64 + 2 * aKoff);
            ((bf16x8*)&af)[1] = *(const bf16x8*)(aBase + s2 * 64 + 2 * aKoff + 32);
#pragma unroll
            for (int j = 0; j < 4; ++j) {
                const char* bp = bBase + j * (16 * LDS_ROW) + s2 * 64 + 2 * bKoff;
                bf16x16 bf_;
                ((bf16x8*)&bf_)[0] = *(const bf16x8*)(bp);
                ((bf16x8*)&bf_)[1] = *(const bf16x8*)(bp + 32);
                acc[j] = __builtin_amdgcn_wmma_f32_16x16x32_bf16(
                    false, af, false, bf_, (short)0, acc[j], false, false);
            }
        }
        __syncthreads();   // protect buffer reuse (stage ks+2 overwrites cur)
    }

    // Epilogue: acc reg r holds (M = mBase + r + 8*(lane>=16), N = nBase+16j+(lane&15))
    float xv[8];
#pragma unroll
    for (int r = 0; r < 8; ++r)
        xv[r] = x_cur[mBase + r + ((lane >> 4) << 3)];

#pragma unroll
    for (int j = 0; j < 4; ++j) {
        const int   n   = nBase + j * 16 + (lane & 15);
        const float bs  = bias_sum[n];
        const float wih = W_ih[n];
#pragma unroll
        for (int r = 0; r < 8; ++r) {
            const int m = mBase + r + ((lane >> 4) << 3);
            const float pre = acc[j][r] + bs + xv[r] * wih;
            h_out[(size_t)m * H_SZ + n] = (__bf16)tanhf(pre);
        }
    }
}

// ---------------------------------------------------------------------------
// Head: y[b] = dot(h[b,:], W_out) + b_out ; out[t,b] = y ;
//       x_next[b] = tf[t] ? tgt[t,b] : y.   One wave per batch row.
// ---------------------------------------------------------------------------
__global__ __launch_bounds__(128) void rnn_step_head(
    const __bf16* __restrict__ h,       // [B,H] bf16 (this step's hidden)
    const float*  __restrict__ W_out,   // [H]
    const float*  __restrict__ b_out,   // [1]
    const float*  __restrict__ tgt_t,   // targets + t*B
    const unsigned char* __restrict__ tf_mask,
    int t,
    float* __restrict__ x_next,         // [B]
    float* __restrict__ y_out)          // d_out + t*B
{
    const int lane = threadIdx.x & 31;
    const int row  = blockIdx.x * 4 + (threadIdx.x >> 5);
    const __bf16* hr = h + (size_t)row * H_SZ;

    float sum = 0.0f;
#pragma unroll 2
    for (int p = 0; p < H_SZ; p += 256) {
        const int k = p + lane * 8;
        const bf16x8 hv = *(const bf16x8*)(hr + k);
        const v4f w0 = *(const v4f*)(W_out + k);
        const v4f w1 = *(const v4f*)(W_out + k + 4);
        sum += (float)hv[0] * w0[0] + (float)hv[1] * w0[1] +
               (float)hv[2] * w0[2] + (float)hv[3] * w0[3] +
               (float)hv[4] * w1[0] + (float)hv[5] * w1[1] +
               (float)hv[6] * w1[2] + (float)hv[7] * w1[3];
    }
#pragma unroll
    for (int off = 16; off > 0; off >>= 1)
        sum += __shfl_down(sum, off, 32);

    if (lane == 0) {
        const float y = sum + b_out[0];
        y_out[row]  = y;
        x_next[row] = tf_mask[t] ? tgt_t[row] : y;
    }
}

// ---------------------------------------------------------------------------
extern "C" void kernel_launch(void* const* d_in, const int* in_sizes, int n_in,
                              void* d_out, int out_size, void* d_ws, size_t ws_size,
                              hipStream_t stream) {
    (void)in_sizes; (void)n_in; (void)out_size; (void)ws_size;

    const float* initial_input = (const float*)d_in[0];   // [B,1]
    const float* hidden        = (const float*)d_in[1];   // [B,H]
    const float* targets       = (const float*)d_in[2];   // [T,B,1]
    const float* W_ih          = (const float*)d_in[3];   // [H,1]
    const float* b_ih          = (const float*)d_in[4];   // [H]
    const float* W_hh          = (const float*)d_in[5];   // [H,H]
    const float* b_hh          = (const float*)d_in[6];   // [H]
    const float* W_out         = (const float*)d_in[7];   // [1,H]
    const float* b_out         = (const float*)d_in[8];   // [1]
    const unsigned char* tf    = (const unsigned char*)d_in[9]; // [T] bool
    float* out = (float*)d_out;                           // [T,B,1]

    // Workspace layout (~12.6 MB total)
    char* ws = (char*)d_ws;
    __bf16* Whh_bf   = (__bf16*)(ws);
    __bf16* hbuf0    = (__bf16*)(ws + 8388608);              // 2 MB
    __bf16* hbuf1    = (__bf16*)(ws + 8388608 + 2097152);    // 2 MB
    float*  x_cur    = (float*)(ws + 8388608 + 2 * 2097152);
    float*  bias_sum = (float*)(ws + 8388608 + 2 * 2097152 + 2048);

    rnn_prep<<<(H_SZ * H_SZ) / 256, 256, 0, stream>>>(
        W_hh, hidden, b_ih, b_hh, initial_input,
        Whh_bf, hbuf0, bias_sum, x_cur);

    const dim3 gemmGrid(H_SZ / 128, B_SZ / 64);  // (16, 8)
    for (int t = 0; t < T_STEPS; ++t) {
        __bf16* hin  = (t & 1) ? hbuf1 : hbuf0;
        __bf16* hout = (t & 1) ? hbuf0 : hbuf1;
        rnn_step_gemm<<<gemmGrid, 256, 0, stream>>>(
            hin, hout, x_cur, Whh_bf, W_ih, bias_sum);
        rnn_step_head<<<B_SZ / 4, 128, 0, stream>>>(
            hout, W_out, b_out, targets + (size_t)t * B_SZ, tf, t,
            x_cur, out + (size_t)t * B_SZ);
    }
}
